// EncoderLayer_52106543235229
// MI455X (gfx1250) — matrix operand order repaired and checked by reading, not compiled, not run on hardware
//
#include <hip/hip_runtime.h>

typedef _Float16 half_t;
typedef __attribute__((ext_vector_type(16))) _Float16 v16h;
typedef __attribute__((ext_vector_type(8)))  float    v8f;

union F16x16 { v16h v; uint4 q[2]; half_t h[16]; };

// ---------------------------------------------------------------------------
// f32 -> f16 convert with zero padding (for FF=500 -> 512 padded tiles)
// ---------------------------------------------------------------------------
__global__ void convert_pad_kernel(const float* __restrict__ src, half_t* __restrict__ dst,
                                   int srcRows, int srcCols, int dstRows, int dstCols) {
    int idx = blockIdx.x * 256 + threadIdx.x;
    int total = dstRows * dstCols;
    if (idx >= total) return;
    int r = idx / dstCols, c = idx - r * dstCols;
    float v = (r < srcRows && c < srcCols) ? src[(size_t)r * srcCols + c] : 0.0f;
    dst[idx] = (half_t)v;
}

// ---------------------------------------------------------------------------
// LayerNorm (torch-style: Bessel std, eps added to std) -> f16 output
// one block (256 threads) per row of 1024 features
// ---------------------------------------------------------------------------
__global__ __launch_bounds__(256) void layernorm_kernel(const float* __restrict__ x,
                                                        const float* __restrict__ alpha,
                                                        const float* __restrict__ beta,
                                                        half_t* __restrict__ out) {
    constexpr int F = 1024;
    __shared__ float s1[256];
    __shared__ float s2[256];
    int row = blockIdx.x;
    const float* xr = x + (size_t)row * F;
    float a = 0.f, b = 0.f;
    float vals[4];
#pragma unroll
    for (int i = 0; i < 4; ++i) {
        float v = xr[threadIdx.x + i * 256];
        vals[i] = v; a += v; b += v * v;
    }
    s1[threadIdx.x] = a; s2[threadIdx.x] = b;
    __syncthreads();
    for (int off = 128; off > 0; off >>= 1) {
        if (threadIdx.x < off) {
            s1[threadIdx.x] += s1[threadIdx.x + off];
            s2[threadIdx.x] += s2[threadIdx.x + off];
        }
        __syncthreads();
    }
    float m   = s1[0] * (1.0f / F);
    float var = (s2[0] - (float)F * m * m) * (1.0f / (F - 1));
    float inv = 1.0f / (sqrtf(fmaxf(var, 0.f)) + 1e-6f);
#pragma unroll
    for (int i = 0; i < 4; ++i) {
        int c = threadIdx.x + i * 256;
        out[(size_t)row * F + c] = (half_t)(alpha[c] * (vals[i] - m) * inv + beta[c]);
    }
}

// ---------------------------------------------------------------------------
// WMMA GEMM: C[M,N] = A[M,K](f16) * B[K,N](f16) + bias, epilogue by MODE
//   MODE 0: store f16
//   MODE 1: ELU then store f16
//   MODE 2: add f32 residual, store f32
// Block: 256 threads (8 waves), tile 128(M) x 64(N), K-step 32.
// Wave w: 32x32 subtile -> 2x2 WMMA 16x16 tiles.
// ---------------------------------------------------------------------------
template <int MODE>
__global__ __launch_bounds__(256) void gemm_kernel(const half_t* __restrict__ A,
                                                   const half_t* __restrict__ Bw,
                                                   const float* __restrict__ bias, int biasN,
                                                   const float* __restrict__ resid,
                                                   float* __restrict__ outF,
                                                   half_t* __restrict__ outH,
                                                   int M, int N, int K) {
    __shared__ half_t As[128 * 40];   // row-major [128][32], stride 40 (80B, 16B-aligned)
    __shared__ half_t Bt[64 * 40];    // transposed [n][k] = [64][32], stride 40
    int tid  = threadIdx.x;
    int lane = tid & 31;
    int w    = tid >> 5;
    int grp  = lane >> 4, ln = lane & 15;
    int wm   = (w & 3) * 32, wn = (w >> 2) * 32;
    int m0   = blockIdx.y * 128, n0 = blockIdx.x * 64;

    v8f c00 = {}, c01 = {}, c10 = {}, c11 = {};

    for (int k0 = 0; k0 < K; k0 += 32) {
        __syncthreads();
        // stage A tile: 128x32 halves as 512 uint4
        for (int u = tid; u < 512; u += 256) {
            int row = u >> 2, c8 = (u & 3) * 8;
            *(uint4*)(&As[row * 40 + c8]) =
                *(const uint4*)(A + (size_t)(m0 + row) * K + k0 + c8);
        }
        // stage B tile transposed: read 8 halves along n, scatter to Bt[n][k]
        {
            int kk = tid >> 3, n8 = (tid & 7) * 8;
            union { uint4 q; half_t h[8]; } t;
            t.q = *(const uint4*)(Bw + (size_t)(k0 + kk) * N + n0 + n8);
#pragma unroll
            for (int j = 0; j < 8; ++j) Bt[(n8 + j) * 40 + kk] = t.h[j];
        }
        __syncthreads();

        F16x16 a0, a1, b0, b1;
        int kb = grp * 8;
        const half_t* ar0 = &As[(wm + ln) * 40 + kb];
        a0.q[0] = *(const uint4*)ar0;        a0.q[1] = *(const uint4*)(ar0 + 16);
        const half_t* ar1 = &As[(wm + 16 + ln) * 40 + kb];
        a1.q[0] = *(const uint4*)ar1;        a1.q[1] = *(const uint4*)(ar1 + 16);
        const half_t* br0 = &Bt[(wn + ln) * 40 + kb];
        b0.q[0] = *(const uint4*)br0;        b0.q[1] = *(const uint4*)(br0 + 16);
        const half_t* br1 = &Bt[(wn + 16 + ln) * 40 + kb];
        b1.q[0] = *(const uint4*)br1;        b1.q[1] = *(const uint4*)(br1 + 16);

        c00 = __builtin_amdgcn_wmma_f32_16x16x32_f16(false, a0.v, false, b0.v, (short)0, c00, false, false);
        c01 = __builtin_amdgcn_wmma_f32_16x16x32_f16(false, a0.v, false, b1.v, (short)0, c01, false, false);
        c10 = __builtin_amdgcn_wmma_f32_16x16x32_f16(false, a1.v, false, b0.v, (short)0, c10, false, false);
        c11 = __builtin_amdgcn_wmma_f32_16x16x32_f16(false, a1.v, false, b1.v, (short)0, c11, false, false);
    }

    int colA = n0 + wn + ln, colB = colA + 16;
    float bA = (colA < biasN) ? bias[colA] : 0.f;
    float bB = (colB < biasN) ? bias[colB] : 0.f;
#pragma unroll
    for (int r = 0; r < 8; ++r) {
        int rowA = m0 + wm + r + grp * 8;
        int rowB = rowA + 16;
        float v0 = c00[r] + bA, v1 = c01[r] + bB, v2 = c10[r] + bA, v3 = c11[r] + bB;
        size_t i0 = (size_t)rowA * N + colA, i1 = (size_t)rowA * N + colB;
        size_t i2 = (size_t)rowB * N + colA, i3 = (size_t)rowB * N + colB;
        if (MODE == 0) {
            outH[i0] = (half_t)v0; outH[i1] = (half_t)v1;
            outH[i2] = (half_t)v2; outH[i3] = (half_t)v3;
        } else if (MODE == 1) {
            v0 = v0 > 0.f ? v0 : expm1f(v0);
            v1 = v1 > 0.f ? v1 : expm1f(v1);
            v2 = v2 > 0.f ? v2 : expm1f(v2);
            v3 = v3 > 0.f ? v3 : expm1f(v3);
            outH[i0] = (half_t)v0; outH[i1] = (half_t)v1;
            outH[i2] = (half_t)v2; outH[i3] = (half_t)v3;
        } else {
            outF[i0] = resid[i0] + v0; outF[i1] = resid[i1] + v1;
            outF[i2] = resid[i2] + v2; outF[i3] = resid[i3] + v3;
        }
    }
}

// ---------------------------------------------------------------------------
// Attention: one wave per (b, h, 16 q-rows). Flash-style single pass,
// running exp-sum (scores ~N(0,1), no max-shift needed), faithful
// mask==0 -> -1e-9 fill. Q/K fragments direct from global (d-contiguous
// matches A/B lane layouts); probs go through LDS to convert C->A layout.
// ---------------------------------------------------------------------------
__global__ __launch_bounds__(32) void attn_kernel(const half_t* __restrict__ Q,
                                                  const half_t* __restrict__ K,
                                                  const half_t* __restrict__ V,
                                                  const int* __restrict__ mask,
                                                  half_t* __restrict__ Ctx) {
    constexpr int S = 2048, F = 1024, HD = 64;
    __shared__ half_t P[16 * 32];
    int lane = threadIdx.x & 31;
    int grp  = lane >> 4, ln = lane & 15;
    int q0 = blockIdx.x * 16;
    int h  = blockIdx.y;
    int b  = blockIdx.z;

    size_t baseQ = ((size_t)b * S + q0 + ln) * F + h * HD;
    F16x16 aq0, aq1;
    aq0.q[0] = *(const uint4*)(Q + baseQ + grp * 8);
    aq0.q[1] = *(const uint4*)(Q + baseQ + grp * 8 + 16);
    aq1.q[0] = *(const uint4*)(Q + baseQ + 32 + grp * 8);
    aq1.q[1] = *(const uint4*)(Q + baseQ + 32 + grp * 8 + 16);

    v8f acc[4] = {{}, {}, {}, {}};
    float Lp[8] = {0.f, 0.f, 0.f, 0.f, 0.f, 0.f, 0.f, 0.f};
    const size_t mrowbase = ((size_t)b * S + q0) * S;

    for (int ks = 0; ks < S; ks += 32) {
#pragma unroll
        for (int t = 0; t < 2; ++t) {
            int kc = ks + t * 16 + ln;
            size_t baseK = ((size_t)b * S + kc) * F + h * HD;
            F16x16 bk0, bk1;
            bk0.q[0] = *(const uint4*)(K + baseK + grp * 8);
            bk0.q[1] = *(const uint4*)(K + baseK + grp * 8 + 16);
            bk1.q[0] = *(const uint4*)(K + baseK + 32 + grp * 8);
            bk1.q[1] = *(const uint4*)(K + baseK + 32 + grp * 8 + 16);
            v8f c = {};
            c = __builtin_amdgcn_wmma_f32_16x16x32_f16(false, aq0.v, false, bk0.v, (short)0, c, false, false);
            c = __builtin_amdgcn_wmma_f32_16x16x32_f16(false, aq1.v, false, bk1.v, (short)0, c, false, false);
#pragma unroll
            for (int r = 0; r < 8; ++r) {
                int row = r + grp * 8;
                float s = c[r] * 0.125f;
                int mv = mask[mrowbase + (size_t)row * S + kc];
                s = (mv == 0) ? -1e-9f : s;
                float e = __expf(s);
                Lp[r] += e;
                P[row * 32 + t * 16 + ln] = (half_t)e;
            }
        }
        __syncthreads();
        F16x16 ap;
        ap.q[0] = *(const uint4*)(&P[ln * 32 + grp * 8]);
        ap.q[1] = *(const uint4*)(&P[ln * 32 + grp * 8 + 16]);
#pragma unroll
        for (int dt = 0; dt < 4; ++dt) {
            F16x16 bv;
#pragma unroll
            for (int i = 0; i < 16; ++i)
                bv.h[i] = V[((size_t)b * S + ks + grp * 8 + (i & 7) + (i >> 3) * 16) * F + h * HD + dt * 16 + ln];
            acc[dt] = __builtin_amdgcn_wmma_f32_16x16x32_f16(false, ap.v, false, bv.v, (short)0, acc[dt], false, false);
        }
        __syncthreads();
    }

#pragma unroll
    for (int r = 0; r < 8; ++r) {
        float v = Lp[r];
        v += __shfl_xor(v, 1);
        v += __shfl_xor(v, 2);
        v += __shfl_xor(v, 4);
        v += __shfl_xor(v, 8);
        Lp[r] = 1.0f / v;
    }
#pragma unroll
    for (int dt = 0; dt < 4; ++dt) {
#pragma unroll
        for (int r = 0; r < 8; ++r) {
            size_t o = ((size_t)b * S + q0 + r + grp * 8) * F + h * HD + dt * 16 + ln;
            Ctx[o] = (half_t)(acc[dt][r] * Lp[r]);
        }
    }
}

// ---------------------------------------------------------------------------
extern "C" void kernel_launch(void* const* d_in, const int* in_sizes, int n_in,
                              void* d_out, int out_size, void* d_ws, size_t ws_size,
                              hipStream_t stream) {
    (void)in_sizes; (void)n_in; (void)out_size; (void)ws_size;
    constexpr int B = 4, S = 2048, F = 1024, H = 16, FF = 500, FFP = 512;
    const int M = B * S;

    const float* x      = (const float*)d_in[0];
    const int*   mask   = (const int*)  d_in[1];
    const float* alpha1 = (const float*)d_in[2];
    const float* bias1  = (const float*)d_in[3];
    const float* alpha2 = (const float*)d_in[4];
    const float* bias2  = (const float*)d_in[5];
    const float* Wq = (const float*)d_in[6];  const float* bq = (const float*)d_in[7];
    const float* Wk = (const float*)d_in[8];  const float* bk = (const float*)d_in[9];
    const float* Wv = (const float*)d_in[10]; const float* bv = (const float*)d_in[11];
    const float* Wo = (const float*)d_in[12]; const float* bo = (const float*)d_in[13];
    const float* W1 = (const float*)d_in[14]; const float* b1 = (const float*)d_in[15];
    const float* W2 = (const float*)d_in[16]; const float* b2 = (const float*)d_in[17];
    float* out = (float*)d_out;

    char* p = (char*)d_ws;
    auto alloc = [&](size_t bytes) -> char* {
        char* r = p;
        p += (bytes + 255) & ~(size_t)255;
        return r;
    };
    half_t* x2h = (half_t*)alloc((size_t)M * F * 2);     // reused as Ctx
    half_t* Qh  = (half_t*)alloc((size_t)M * F * 2);     // reused as x2b
    half_t* Kh  = (half_t*)alloc((size_t)M * F * 2);     // reused as H buffer
    half_t* Vh  = (half_t*)alloc((size_t)M * F * 2);
    half_t* Wqh = (half_t*)alloc((size_t)F * F * 2);
    half_t* Wkh = (half_t*)alloc((size_t)F * F * 2);
    half_t* Wvh = (half_t*)alloc((size_t)F * F * 2);
    half_t* Woh = (half_t*)alloc((size_t)F * F * 2);
    half_t* W1h = (half_t*)alloc((size_t)F * FFP * 2);
    half_t* W2h = (half_t*)alloc((size_t)FFP * F * 2);
    float*  xres = (float*)alloc((size_t)M * F * 4);

    half_t* Ctx = x2h;   // x2h dead after V projection
    half_t* x2b = Qh;    // Qh dead after attention
    half_t* Hb  = Kh;    // Kh dead after attention

    // weight conversions (W1/W2 zero-padded to 512)
    convert_pad_kernel<<<(F * F + 255) / 256, 256, 0, stream>>>(Wq, Wqh, F, F, F, F);
    convert_pad_kernel<<<(F * F + 255) / 256, 256, 0, stream>>>(Wk, Wkh, F, F, F, F);
    convert_pad_kernel<<<(F * F + 255) / 256, 256, 0, stream>>>(Wv, Wvh, F, F, F, F);
    convert_pad_kernel<<<(F * F + 255) / 256, 256, 0, stream>>>(Wo, Woh, F, F, F, F);
    convert_pad_kernel<<<(F * FFP + 255) / 256, 256, 0, stream>>>(W1, W1h, F, FF, F, FFP);
    convert_pad_kernel<<<(FFP * F + 255) / 256, 256, 0, stream>>>(W2, W2h, FF, F, FFP, F);

    // x2 = norm(x)
    layernorm_kernel<<<M, 256, 0, stream>>>(x, alpha1, bias1, x2h);

    // Q/K/V projections
    dim3 gb(F / 64, M / 128);
    gemm_kernel<0><<<gb, 256, 0, stream>>>(x2h, Wqh, bq, F, nullptr, nullptr, Qh, M, F, F);
    gemm_kernel<0><<<gb, 256, 0, stream>>>(x2h, Wkh, bk, F, nullptr, nullptr, Kh, M, F, F);
    gemm_kernel<0><<<gb, 256, 0, stream>>>(x2h, Wvh, bv, F, nullptr, nullptr, Vh, M, F, F);

    // attention -> Ctx (f16)
    dim3 ga(S / 16, H, B);
    attn_kernel<<<ga, 32, 0, stream>>>(Qh, Kh, Vh, mask, Ctx);

    // x = x + Ctx @ Wo + bo
    gemm_kernel<2><<<gb, 256, 0, stream>>>(Ctx, Woh, bo, F, x, xres, nullptr, M, F, F);

    // x2 = norm(x)
    layernorm_kernel<<<M, 256, 0, stream>>>(xres, alpha2, bias2, x2b);

    // h = elu(x2 @ W1 + b1)   (N padded to 512; pad cols give elu(0)=0)
    dim3 g1(FFP / 64, M / 128);
    gemm_kernel<1><<<g1, 256, 0, stream>>>(x2b, W1h, b1, FF, nullptr, nullptr, Hb, M, FFP, F);

    // out = x + h @ W2 + b2   (pad rows of W2 are zero -> no contribution)
    gemm_kernel<2><<<gb, 256, 0, stream>>>(Hb, W2h, b2, F, xres, out, nullptr, M, F, FFP);
}
